// LSTM_78640851190007
// MI455X (gfx1250) — matrix operand order repaired and checked by reading, not compiled, not run on hardware
//
#include <hip/hip_runtime.h>
#include <hip/hip_bf16.h>

// ---------------------------------------------------------------------------
// 3-layer LSTM (H=1024, T=4096, batch=1) persistent-kernel for gfx1250.
//   - bf16 weights (40 MB) resident in the 192 MB L2; f32 accumulation via
//     v_wmma_f32_16x16x32_bf16.
//   - 192 workgroups: 64 per layer (one 16-unit hidden block each), 4 waves
//     per WG = 4 K-quarters, LDS reduction -> 256 KB of L2 weight traffic per
//     WGP per timestep.
//   - Diagonal layer pipeline: global step s runs layer l at t = s - l;
//     one grid barrier per step (T+2 total), h double-buffered by t parity,
//     cell state c lives in registers for the whole sequence.
//   - Branch-free v_exp_f32/v_rcp_f32 activations on the serial critical path.
// ---------------------------------------------------------------------------

#define HSZ   1024
#define TSEQ  4096
#define NLWG  64            // workgroups per layer (one hidden block each)
#define NWG   (3 * NLWG)    // 192 total workgroups
#define TPB   128           // 4 waves of 32 (wave = K-quarter)

typedef __attribute__((ext_vector_type(16))) __bf16 v16bf;
typedef __attribute__((ext_vector_type(8)))  __bf16 v8bf;
typedef __attribute__((ext_vector_type(8)))  float  v8f;

__device__ __forceinline__ unsigned short f2bf(float f) {
    unsigned u = __builtin_bit_cast(unsigned, f);
    unsigned r = (u + 0x7FFFu + ((u >> 16) & 1u)) >> 16;
    return (unsigned short)r;
}

// Branch-free activations: v_exp_f32 + v_rcp_f32 only (short serial chain).
__device__ __forceinline__ float fast_sigmoid(float x) {
    float e = __builtin_amdgcn_exp2f(-1.4426950408889634f * x);
    return __builtin_amdgcn_rcpf(1.0f + e);
}
__device__ __forceinline__ float fast_tanh(float x) {
    float e = __builtin_amdgcn_exp2f(-2.8853900817779268f * x);
    return 2.0f * __builtin_amdgcn_rcpf(1.0f + e) - 1.0f;
}

// ---------------------------------------------------------------------------
// Init: fp32 -> bf16 weight reorg into [W_ih | W_hh] row-major blocks, fused
// biases, zero h buffers / head accumulator / barrier counter (every call, so
// graph replay is deterministic).
// ---------------------------------------------------------------------------
__global__ void lstm_init_kernel(
    const float* __restrict__ Whh0,
    const float* __restrict__ bih0,  const float* __restrict__ bhh0,
    const float* __restrict__ Wih12, const float* __restrict__ Whh12,
    const float* __restrict__ bih12, const float* __restrict__ bhh12,
    unsigned short* __restrict__ wc0,
    unsigned short* __restrict__ wc1,
    unsigned short* __restrict__ wc2,
    float* __restrict__ bias,          // [3][4096]
    unsigned short* __restrict__ hbuf, // [3][2][HSZ]
    float* __restrict__ yacc,          // [TSEQ]
    unsigned int* __restrict__ cnt)
{
    const long long N = 4096LL * 2048LL;
    long long stride = (long long)gridDim.x * blockDim.x;
    for (long long i = (long long)blockIdx.x * blockDim.x + threadIdx.x; i < N; i += stride) {
        int r = (int)(i >> 11);          // gate row 0..4095
        int k = (int)(i & 2047);         // combined K 0..2047
        float s1 = (k < HSZ) ? Wih12[r * HSZ + k] : Whh12[r * HSZ + (k - HSZ)];
        float s2 = (k < HSZ) ? Wih12[4096 * HSZ + r * HSZ + k]
                             : Whh12[4096 * HSZ + r * HSZ + (k - HSZ)];
        wc1[i] = f2bf(s1);
        wc2[i] = f2bf(s2);
        if (k < HSZ) wc0[(long long)r * HSZ + k] = f2bf(Whh0[(long long)r * HSZ + k]);
        if (i < 4096) {
            int j = (int)i;
            bias[j]        = bih0[j] + bhh0[j];
            bias[4096 + j] = bih12[j] + bhh12[j];
            bias[8192 + j] = bih12[4096 + j] + bhh12[4096 + j];
            yacc[j] = 0.0f;
        }
        if (i < 3 * 2 * HSZ) hbuf[i] = 0;
        if (i == 0) *cnt = 0u;
    }
}

// ---------------------------------------------------------------------------
// Persistent pipelined LSTM kernel.
// ---------------------------------------------------------------------------
__global__ __launch_bounds__(TPB, 1) void lstm_persist_kernel(
    const float* __restrict__ x,      // (1,T,2)
    const float* __restrict__ Wih0,   // (4096,2) fp32: x-term done scalar
    const float* __restrict__ Wres,   // (1,H)
    const unsigned short* __restrict__ wc0,  // [4096][1024]  bf16
    const unsigned short* __restrict__ wc1,  // [4096][2048]  bf16
    const unsigned short* __restrict__ wc2,  // [4096][2048]  bf16
    const float* __restrict__ bias,          // [3][4096]
    unsigned short* __restrict__ hbuf,       // [3][2][HSZ] bf16
    float* __restrict__ yacc,                // [TSEQ]
    unsigned int* __restrict__ cnt)
{
    __shared__ __align__(16) unsigned short sV[2 * HSZ];   // staged input vectors
    __shared__ __align__(16) float redf[3 * 32 * 32];      // K-quarter partials

    const int tid   = threadIdx.x;
    const int lane  = tid & 31;
    const int q     = tid >> 5;          // K-quarter 0..3
    const int wg    = blockIdx.x;
    const int layer = wg / NLWG;         // 0..2
    const int lwg   = wg - layer * NLWG; // 0..63
    const int ub    = lwg * 16;          // hidden base index of this block
    const int half  = lane >> 4;         // 0 or 1
    const int m     = lane & 15;         // WMMA A row owned by this lane
    const int r0    = half * 8;          // D rows held by this lane

    const int KL = (layer == 0) ? HSZ : 2 * HSZ;
    const int Kq = KL >> 2;              // K per wave
    const int kb = q * Kq;
    const unsigned short* W = (layer == 0) ? wc0 : ((layer == 1) ? wc1 : wc2);
    const float* bl = bias + layer * 4096;

    float c_reg[8];                      // cell state (only wave 0's is used)
    #pragma unroll
    for (int r = 0; r < 8; ++r) c_reg[r] = 0.0f;

    for (int s = 0; s < TSEQ + 2; ++s) {
        const int t = s - layer;
        const bool active = (t >= 0) && (t < TSEQ);
        if (active) {
            // ---- stage input vectors into LDS (bf16) -----------------------
            // layer 0 : sV[0:H)  = h0(t-1)
            // layer 1+: sV[0:H)  = h_{l-1}(t), sV[H:2H) = h_l(t-1)
            if (layer == 0) {
                const uint4* s0 = (const uint4*)(hbuf + (0 * 2 + ((t + 1) & 1)) * HSZ);
                ((uint4*)sV)[tid] = s0[tid];                 // 128 * 16B = 2KB
            } else {
                const uint4* s0 = (const uint4*)(hbuf + ((layer - 1) * 2 + (t & 1)) * HSZ);
                const uint4* s1 = (const uint4*)(hbuf + (layer * 2 + ((t + 1) & 1)) * HSZ);
                ((uint4*)sV)[tid]         = s0[tid];
                ((uint4*)(sV + HSZ))[tid] = s1[tid];
            }
            __syncthreads();

            // ---- K-quarter WMMA accumulation -------------------------------
            v8f acc[4] = {};             // i, f, g, o partials
            const unsigned short* Arow = W + (size_t)(ub + m) * KL + half * 8;
            for (int k0 = kb; k0 < kb + Kq; k0 += 32) {
                // B (32x16), vector replicated across all 16 columns:
                // lanes 0-15 take vec[k0+{0..7,16..23}], lanes 16-31 vec[k0+{8..15,24..31}]
                const v8bf b_lo = *(const v8bf*)(sV + k0 + half * 8);
                const v8bf b_hi = *(const v8bf*)(sV + k0 + half * 8 + 16);
                const v16bf bmat = __builtin_shufflevector(
                    b_lo, b_hi, 0,1,2,3,4,5,6,7,8,9,10,11,12,13,14,15);
                #pragma unroll
                for (int g = 0; g < 4; ++g) {
                    // A (16x32): lane<16 -> row m, K {0..7,16..23};
                    //            lane>=16 -> row m, K {8..15,24..31}
                    const unsigned short* ap = Arow + (size_t)g * HSZ * KL + k0;
                    const v8bf a_lo = *(const v8bf*)(ap);
                    const v8bf a_hi = *(const v8bf*)(ap + 16);
                    const v16bf amat = __builtin_shufflevector(
                        a_lo, a_hi, 0,1,2,3,4,5,6,7,8,9,10,11,12,13,14,15);
                    acc[g] = __builtin_amdgcn_wmma_f32_16x16x32_bf16(
                        false, amat, false, bmat, (short)0, acc[g], false, false);
                }
            }

            // ---- cross-wave K reduction through LDS ------------------------
            if (q != 0) {
                float* dst = redf + ((size_t)(q - 1) * 32 + lane) * 32;
                #pragma unroll
                for (int g = 0; g < 4; ++g) *(v8f*)(dst + g * 8) = acc[g];
            }
            __syncthreads();

            if (q == 0) {
                #pragma unroll
                for (int w = 0; w < 3; ++w) {
                    const float* src = redf + ((size_t)w * 32 + lane) * 32;
                    #pragma unroll
                    for (int g = 0; g < 4; ++g)
                        acc[g] = acc[g] + *(const v8f*)(src + g * 8);
                }

                // bias (+ layer0 x-term)
                union { v8f v; float f[8]; } ua[4];
                #pragma unroll
                for (int g = 0; g < 4; ++g) {
                    ua[g].v = acc[g];
                    const float* bp = bl + g * HSZ + ub + r0;
                    #pragma unroll
                    for (int r = 0; r < 8; ++r) ua[g].f[r] += bp[r];
                }
                if (layer == 0) {
                    const float xv0 = x[2 * t], xv1 = x[2 * t + 1];
                    #pragma unroll
                    for (int g = 0; g < 4; ++g) {
                        const float* wp = Wih0 + (size_t)(g * HSZ + ub + r0) * 2;
                        #pragma unroll
                        for (int r = 0; r < 8; ++r)
                            ua[g].f[r] += wp[2 * r] * xv0 + wp[2 * r + 1] * xv1;
                    }
                }

                // elementwise gate math (branch-free); lane holds rows r0..r0+7
                __align__(16) unsigned short hout[8];
                float hval[8];
                #pragma unroll
                for (int r = 0; r < 8; ++r) {
                    float iv = fast_sigmoid(ua[0].f[r]);
                    float fv = fast_sigmoid(ua[1].f[r]);
                    float gv = fast_tanh(ua[2].f[r]);
                    float ov = fast_sigmoid(ua[3].f[r]);
                    float cn = fv * c_reg[r] + iv * gv;
                    c_reg[r] = cn;
                    float hv = ov * fast_tanh(cn);
                    hval[r]  = hv;
                    hout[r]  = f2bf(hv);
                }
                if (lane == 0 || lane == 16) {
                    unsigned short* hd = hbuf + (layer * 2 + (t & 1)) * HSZ + ub + r0;
                    *(uint4*)hd = *(const uint4*)hout;
                    if (layer == 2) {            // fused output head partial dot
                        float p = 0.0f;
                        const float* wr = Wres + ub + r0;
                        #pragma unroll
                        for (int r = 0; r < 8; ++r) p += wr[r] * hval[r];
                        atomicAdd(&yacc[t], p);
                    }
                }
            }
        }

        // ---- grid-wide step barrier (monotonic counter, no reset race) -----
        __syncthreads();
        if (tid == 0) {
            __threadfence();
            atomicAdd(cnt, 1u);
            const unsigned int target = (unsigned int)(s + 1) * (unsigned int)NWG;
            while (__hip_atomic_load(cnt, __ATOMIC_ACQUIRE, __HIP_MEMORY_SCOPE_AGENT) < target) {
                __builtin_amdgcn_s_sleep(1);
            }
        }
        __syncthreads();
        __threadfence();
    }
}

// ---------------------------------------------------------------------------
// Head tail: y[t] = sigmoid(acc[t] + b_res)
// ---------------------------------------------------------------------------
__global__ void lstm_head_kernel(const float* __restrict__ yacc,
                                 const float* __restrict__ b_res,
                                 float* __restrict__ out)
{
    int i = blockIdx.x * blockDim.x + threadIdx.x;
    if (i < TSEQ) out[i] = fast_sigmoid(yacc[i] + b_res[0]);
}

// ---------------------------------------------------------------------------
extern "C" void kernel_launch(void* const* d_in, const int* in_sizes, int n_in,
                              void* d_out, int out_size, void* d_ws, size_t ws_size,
                              hipStream_t stream)
{
    (void)in_sizes; (void)n_in; (void)out_size; (void)ws_size;

    const float* x     = (const float*)d_in[0];
    const float* Wih0  = (const float*)d_in[1];
    const float* Whh0  = (const float*)d_in[2];
    const float* bih0  = (const float*)d_in[3];
    const float* bhh0  = (const float*)d_in[4];
    const float* Wih12 = (const float*)d_in[5];
    const float* Whh12 = (const float*)d_in[6];
    const float* bih12 = (const float*)d_in[7];
    const float* bhh12 = (const float*)d_in[8];
    const float* Wres  = (const float*)d_in[9];
    const float* bres  = (const float*)d_in[10];
    float* out = (float*)d_out;

    // workspace carve-up
    char* ws = (char*)d_ws;
    size_t o = 0;
    unsigned short* wc0 = (unsigned short*)(ws + o); o += (size_t)4096 * 1024 * 2; // 8 MiB
    unsigned short* wc1 = (unsigned short*)(ws + o); o += (size_t)4096 * 2048 * 2; // 16 MiB
    unsigned short* wc2 = (unsigned short*)(ws + o); o += (size_t)4096 * 2048 * 2; // 16 MiB
    float* bias          = (float*)(ws + o);          o += (size_t)3 * 4096 * 4;
    unsigned short* hbuf = (unsigned short*)(ws + o); o += (size_t)3 * 2 * HSZ * 2;
    o = (o + 255) & ~(size_t)255;
    float* yacc          = (float*)(ws + o);          o += (size_t)TSEQ * 4;
    o = (o + 255) & ~(size_t)255;
    unsigned int* cnt    = (unsigned int*)(ws + o);   o += 256;

    lstm_init_kernel<<<4096, 256, 0, stream>>>(
        Whh0, bih0, bhh0, Wih12, Whh12, bih12, bhh12,
        wc0, wc1, wc2, bias, hbuf, yacc, cnt);

    lstm_persist_kernel<<<NWG, TPB, 0, stream>>>(
        x, Wih0, Wres, wc0, wc1, wc2, bias, hbuf, yacc, cnt);

    lstm_head_kernel<<<(TSEQ + 255) / 256, 256, 0, stream>>>(yacc, bres, out);
}
